// TranslateModel_90658169684197
// MI455X (gfx1250) — matrix operand order repaired and checked by reading, not compiled, not run on hardware
//
#include <hip/hip_runtime.h>
#include <hip/hip_bf16.h>
#include <math.h>

// ---------------- model constants ----------------
constexpr int kV  = 32000;
constexpr int kD  = 1024;
constexpr int kNL = 4;
constexpr int kDI = 2048;
constexpr int kDS = 8;
constexpr int kDC = 4;
constexpr int kDR = 64;
constexpr int kB  = 2;
constexpr int kL  = 2048;
constexpr int kM  = kB * kL;   // 4096 token rows

typedef __attribute__((ext_vector_type(16))) __bf16 bf16x16;
typedef __attribute__((ext_vector_type(4)))  __bf16 bf16x4;
typedef __attribute__((ext_vector_type(8)))  float  floatx8;

__device__ __forceinline__ float silu_f(float v) {
    return v / (1.f + __expf(-v));
}

// =====================================================================
// f32 -> bf16 bulk conversion (weights converted once per forward)
// n must be a multiple of 4 (true for all uses).
// =====================================================================
__global__ __launch_bounds__(256)
void cvt_bf16_kernel(const float* __restrict__ src, __bf16* __restrict__ dst)
{
    const size_t i = ((size_t)blockIdx.x * blockDim.x + threadIdx.x) * 4;
    const float4 v = *(const float4*)(src + i);
    bf16x4 o;
    o.x = (__bf16)v.x; o.y = (__bf16)v.y; o.z = (__bf16)v.z; o.w = (__bf16)v.w;
    *(bf16x4*)(dst + i) = o;
}

// =====================================================================
// GEMM: C[M,N] = act(A[M,K] @ W[N,K]^T + bias[N]),  A/W are bf16.
// 256 threads = 8 waves; block tile 128(M) x 64(N); K-step 32.
// Double-buffered LDS filled with global_load_async_to_lds_b128
// (ASYNCcnt pipeline), v_wmma_f32_16x16x32_bf16 core, 2x2 WMMA/wave.
// Requires: M % 128 == 0, K % 32 == 0 (true for all uses). N may be ragged.
// =====================================================================
#define TSB 80   // LDS row stride in bytes (40 bf16, 16B aligned)

__global__ __launch_bounds__(256)
void gemm_bf16_wmma(const __bf16* __restrict__ A, const __bf16* __restrict__ W,
                    const float* __restrict__ bias, float* __restrict__ C,
                    __bf16* __restrict__ Cbf,
                    int N, int K, int lda, int ldw, int ldc, int act)
{
    __shared__ __align__(16) __bf16 As[2][128 * (TSB / 2)];
    __shared__ __align__(16) __bf16 Bs[2][ 64 * (TSB / 2)];

    const int m0   = blockIdx.y * 128;
    const int n0   = blockIdx.x * 64;
    const int tid  = threadIdx.x;
    const int lane = tid & 31;
    const int wave = tid >> 5;
    const int wm   = (wave >> 1) * 32;      // wave M offset: 0/32/64/96
    const int wn   = (wave & 1) * 32;       // wave N offset: 0/32
    const int hv   = lane >> 4;             // lane half (0/1)
    const int l16  = lane & 15;

    const unsigned asBase = (unsigned)(size_t)(&As[0][0]);
    const unsigned bsBase = (unsigned)(size_t)(&Bs[0][0]);

    // issue async loads for one K-step tile into LDS buffer `buf`
    auto stage = [&](int buf, int kk) {
        // A tile: 128 rows x 32 bf16 (64B) = 512 x 16B chunks, 2 per thread
        #pragma unroll
        for (int p = 0; p < 2; ++p) {
            const int i  = tid + p * 256;
            const int r  = i >> 2;
            const int ch = i & 3;
            const __bf16* g = A + (size_t)(m0 + r) * lda + kk + ch * 8;
            const unsigned l = asBase + (unsigned)buf * (128 * TSB) + r * TSB + ch * 16;
            asm volatile("global_load_async_to_lds_b128 %0, %1, off"
                         :: "v"(l), "v"(g) : "memory");
        }
        // B tile: 64 rows x 32 bf16 = 256 x 16B chunks, 1 per thread.
        // Ragged N: clamp row (garbage rows only feed masked output columns).
        {
            const int r  = tid >> 2;
            const int ch = tid & 3;
            const int n  = n0 + r;
            const int rn = (n < N) ? n : (N - 1);
            const __bf16* g = W + (size_t)rn * ldw + kk + ch * 8;
            const unsigned l = bsBase + (unsigned)buf * (64 * TSB) + r * TSB + ch * 16;
            asm volatile("global_load_async_to_lds_b128 %0, %1, off"
                         :: "v"(l), "v"(g) : "memory");
        }
    };

    floatx8 acc[2][2] = {};

    const int T = K >> 5;                   // number of 32-wide K steps
    stage(0, 0);
    for (int t = 0; t < T; ++t) {
        const int cur = t & 1;
        if (t + 1 < T) {
            stage(cur ^ 1, (t + 1) * 32);
            asm volatile("s_wait_asynccnt 0x3" ::: "memory"); // tile t landed
        } else {
            asm volatile("s_wait_asynccnt 0x0" ::: "memory");
        }
        __syncthreads();

        // fragments per the 16-bit A(16x32) / B(32x16) lane layouts
        bf16x16 afrag[2], bfrag[2];
        #pragma unroll
        for (int i = 0; i < 2; ++i) {
            const __bf16* ap = &As[cur][(wm + i * 16 + l16) * (TSB / 2)];
            #pragma unroll
            for (int q = 0; q < 8; ++q) {
                afrag[i][q]     = ap[hv * 8 + q];       // K = hv*8 + q
                afrag[i][8 + q] = ap[16 + hv * 8 + q];  // K = 16 + hv*8 + q
            }
        }
        #pragma unroll
        for (int j = 0; j < 2; ++j) {
            const __bf16* bp = &Bs[cur][(wn + j * 16 + l16) * (TSB / 2)];
            #pragma unroll
            for (int q = 0; q < 8; ++q) {
                bfrag[j][q]     = bp[hv * 8 + q];       // K = hv*8 + q
                bfrag[j][8 + q] = bp[16 + hv * 8 + q];  // K = 16 + hv*8 + q
            }
        }

        #pragma unroll
        for (int i = 0; i < 2; ++i)
            #pragma unroll
            for (int j = 0; j < 2; ++j)
                acc[i][j] = __builtin_amdgcn_wmma_f32_16x16x32_bf16(
                    false, afrag[i], false, bfrag[j],
                    (short)0, acc[i][j], false, false);
        __syncthreads();   // all waves done reading `cur` before it is refilled
    }

    // ---- epilogue: bias + activation + f32 store (+ optional bf16 copy) ----
    #pragma unroll
    for (int i = 0; i < 2; ++i) {
        #pragma unroll
        for (int j = 0; j < 2; ++j) {
            const int col = n0 + wn + j * 16 + l16;
            if (col >= N) continue;
            const float bv = bias ? bias[col] : 0.f;
            #pragma unroll
            for (int r = 0; r < 8; ++r) {
                const int row = m0 + wm + i * 16 + hv * 8 + r;
                float v = acc[i][j][r] + bv;
                if (act == 1) v = (v > 20.f) ? v : log1pf(__expf(v)); // softplus
                const size_t o = (size_t)row * ldc + col;
                C[o] = v;
                if (Cbf) Cbf[o] = (__bf16)v;
            }
        }
    }
}

// =====================================================================
// Embedding gather: x[tok,:] = emb[ids[tok],:]  (f32 + bf16 copies)
// =====================================================================
__global__ __launch_bounds__(256)
void embed_kernel(const int* __restrict__ ids, const float* __restrict__ emb,
                  float* __restrict__ x, __bf16* __restrict__ xbf)
{
    const int tok = blockIdx.x;
    const int id  = ids[tok];
    const float4* src = (const float4*)(emb + (size_t)id * kD);
    float4*       dst = (float4*)(x + (size_t)tok * kD);
    __bf16*       db  = xbf + (size_t)tok * kD;
    for (int c = threadIdx.x; c < kD / 4; c += blockDim.x) {
        const float4 v = src[c];
        dst[c] = v;
        bf16x4 o;
        o.x = (__bf16)v.x; o.y = (__bf16)v.y; o.z = (__bf16)v.z; o.w = (__bf16)v.w;
        *(bf16x4*)(db + c * 4) = o;
    }
}

// =====================================================================
// Causal depthwise conv (DC=4) over first half of xz, + bias + SiLU
// =====================================================================
__global__ __launch_bounds__(256)
void conv_silu_kernel(const float* __restrict__ xz, const float* __restrict__ cw,
                      const float* __restrict__ cb, float* __restrict__ xi,
                      __bf16* __restrict__ xibf)
{
    const size_t i = (size_t)blockIdx.x * blockDim.x + threadIdx.x;
    const int c   = (int)(i % kDI);
    const size_t row = i / kDI;           // b*L + l
    const int l   = (int)(row % kL);
    float acc = cb[c];
    #pragma unroll
    for (int k = 0; k < kDC; ++k) {
        const int d = k - (kDC - 1);      // -3..0
        if (l + d >= 0)
            acc += cw[c * kDC + k] * xz[(row + d) * (2 * kDI) + c];
    }
    const float v = silu_f(acc);
    xi[i]   = v;
    xibf[i] = (__bf16)v;
}

// =====================================================================
// Selective scan: 8 lanes per (b,d) channel (one per state n),
// shuffle-reduce over DS=8. wave32: xor masks 1,2,4 stay in-group.
// =====================================================================
__global__ __launch_bounds__(256)
void scan_kernel(const float* __restrict__ delta, const float* __restrict__ xi,
                 const float* __restrict__ dbl, const float* __restrict__ A_log,
                 float* __restrict__ ys)
{
    const int gid = blockIdx.x * 32 + (threadIdx.x >> 3);  // (b,d) channel
    const int n   = threadIdx.x & 7;
    const int b   = gid / kDI;
    const int d   = gid % kDI;
    const float Ac = -__expf(A_log[d * kDS + n]);
    float h = 0.f;
    const size_t chanBase = (size_t)b * kL * kDI + d;
    const size_t rowBase  = (size_t)b * kL * (kDR + 2 * kDS);
    for (int l = 0; l < kL; ++l) {
        const size_t idx = chanBase + (size_t)l * kDI;
        const float dlt  = delta[idx];
        const float xv   = xi[idx];
        const float* prow = dbl + rowBase + (size_t)l * (kDR + 2 * kDS);
        const float Bv = prow[kDR + n];
        const float Cv = prow[kDR + kDS + n];
        h = __expf(dlt * Ac) * h + dlt * Bv * xv;
        float p = h * Cv;
        p += __shfl_xor(p, 1, 32);
        p += __shfl_xor(p, 2, 32);
        p += __shfl_xor(p, 4, 32);
        if (n == 0) ys[idx] = p;
    }
}

// =====================================================================
// Gate: ys = (ys + xi*Dp) * silu(z),  z = second half of xz. + bf16 copy
// =====================================================================
__global__ __launch_bounds__(256)
void gate_kernel(float* __restrict__ ys, __bf16* __restrict__ ysbf,
                 const float* __restrict__ xi, const float* __restrict__ xz,
                 const float* __restrict__ Dp)
{
    const size_t i = (size_t)blockIdx.x * blockDim.x + threadIdx.x;
    const int c = (int)(i % kDI);
    const size_t row = i / kDI;
    const float z = xz[row * (2 * kDI) + kDI + c];
    const float y = (ys[i] + xi[i] * Dp[c]) * silu_f(z);
    ys[i]   = y;
    ysbf[i] = (__bf16)y;
}

// =====================================================================
// LayerNorm: out = LN(a [+ res]) * g + b ;  f32 and/or bf16 outputs.
// One block per row, D=1024, 256 threads.
// =====================================================================
__global__ __launch_bounds__(256)
void ln_kernel(const float* __restrict__ a, const float* __restrict__ res,
               const float* __restrict__ g, const float* __restrict__ bta,
               float* __restrict__ dst, __bf16* __restrict__ dstbf)
{
    __shared__ float sred[8];
    const int row = blockIdx.x;
    const int tid = threadIdx.x;
    const size_t base = (size_t)row * kD;
    float loc[4];
    float s = 0.f;
    #pragma unroll
    for (int t = 0; t < 4; ++t) {
        const int c = tid + t * 256;
        float v = a[base + c];
        if (res) v += res[base + c];
        loc[t] = v; s += v;
    }
    #pragma unroll
    for (int o = 16; o > 0; o >>= 1) s += __shfl_xor(s, o, 32);
    if ((tid & 31) == 0) sred[tid >> 5] = s;
    __syncthreads();
    float tot = 0.f;
    #pragma unroll
    for (int w = 0; w < 8; ++w) tot += sred[w];
    const float mu = tot / kD;
    __syncthreads();

    float sq = 0.f;
    #pragma unroll
    for (int t = 0; t < 4; ++t) { const float dv = loc[t] - mu; sq += dv * dv; }
    #pragma unroll
    for (int o = 16; o > 0; o >>= 1) sq += __shfl_xor(sq, o, 32);
    if ((tid & 31) == 0) sred[tid >> 5] = sq;
    __syncthreads();
    float tot2 = 0.f;
    #pragma unroll
    for (int w = 0; w < 8; ++w) tot2 += sred[w];
    const float rstd = rsqrtf(tot2 / kD + 1e-5f);
    #pragma unroll
    for (int t = 0; t < 4; ++t) {
        const int c = tid + t * 256;
        const float v = (loc[t] - mu) * rstd * g[c] + bta[c];
        if (dst)   dst[base + c]   = v;
        if (dstbf) dstbf[base + c] = (__bf16)v;
    }
}

// =====================================================================
// Host-side orchestration
// =====================================================================
extern "C" void kernel_launch(void* const* d_in, const int* in_sizes, int n_in,
                              void* d_out, int out_size, void* d_ws, size_t ws_size,
                              hipStream_t stream)
{
    (void)in_sizes; (void)n_in; (void)out_size; (void)ws_size;

    const int*   ids     = (const int*)  d_in[0];
    const float* emb     = (const float*)d_in[1];
    const float* in_w    = (const float*)d_in[2];
    const float* in_b    = (const float*)d_in[3];
    const float* conv_w  = (const float*)d_in[4];
    const float* conv_b  = (const float*)d_in[5];
    const float* xproj_w = (const float*)d_in[6];
    const float* dt_w    = (const float*)d_in[7];
    const float* dt_b    = (const float*)d_in[8];
    const float* A_log   = (const float*)d_in[9];
    const float* Dp      = (const float*)d_in[10];
    const float* out_w   = (const float*)d_in[11];
    const float* out_b   = (const float*)d_in[12];
    const float* ln_g    = (const float*)d_in[13];
    const float* ln_b    = (const float*)d_in[14];
    const float* fn_g    = (const float*)d_in[15];
    const float* fn_b    = (const float*)d_in[16];
    const float* fc_w    = (const float*)d_in[17];
    const float* fc_b    = (const float*)d_in[18];
    float* logits = (float*)d_out;

    // ---------------- workspace layout ----------------
    float* ws    = (float*)d_ws;
    float* x     = ws;                                   // kM*kD
    float* xz    = x     + (size_t)kM * kD;              // kM*2*kDI
    float* xi    = xz    + (size_t)kM * 2 * kDI;         // kM*kDI
    float* dbl   = xi    + (size_t)kM * kDI;             // kM*80
    float* delta = dbl   + (size_t)kM * (kDR + 2 * kDS); // kM*kDI
    float* ysb   = delta + (size_t)kM * kDI;             // kM*kDI
    float* outb  = ysb   + (size_t)kM * kDI;             // kM*kD

    __bf16* bf      = (__bf16*)(outb + (size_t)kM * kD);
    __bf16* x_bf    = bf;                                  // kM*kD
    __bf16* xi_bf   = x_bf    + (size_t)kM * kD;           // kM*kDI
    __bf16* dbl_bf  = xi_bf   + (size_t)kM * kDI;          // kM*80
    __bf16* ys_bf   = dbl_bf  + (size_t)kM * (kDR + 2 * kDS); // kM*kDI
    __bf16* xn_bf   = ys_bf   + (size_t)kM * kDI;          // kM*kD
    __bf16* in_w_bf = xn_bf   + (size_t)kM * kD;           // kNL*2*kDI*kD
    __bf16* xp_w_bf = in_w_bf + (size_t)kNL * 2 * kDI * kD;
    __bf16* dt_w_bf = xp_w_bf + (size_t)kNL * (kDR + 2 * kDS) * kDI;
    __bf16* ou_w_bf = dt_w_bf + (size_t)kNL * kDI * kDR;
    __bf16* fc_w_bf = ou_w_bf + (size_t)kNL * kD * kDI;

    const dim3 blk256(256);
    const size_t nEl = (size_t)kM * kDI;                 // 8.4M
    const dim3 gEl((unsigned)(nEl / 256));

    // ---------------- weight conversion (once per forward) ----------------
    cvt_bf16_kernel<<<dim3((unsigned)((size_t)kNL * 2 * kDI * kD / 1024)), blk256, 0, stream>>>(in_w, in_w_bf);
    cvt_bf16_kernel<<<dim3((unsigned)((size_t)kNL * (kDR + 2 * kDS) * kDI / 1024)), blk256, 0, stream>>>(xproj_w, xp_w_bf);
    cvt_bf16_kernel<<<dim3((unsigned)((size_t)kNL * kDI * kDR / 1024)), blk256, 0, stream>>>(dt_w, dt_w_bf);
    cvt_bf16_kernel<<<dim3((unsigned)((size_t)kNL * kD * kDI / 1024)), blk256, 0, stream>>>(out_w, ou_w_bf);
    cvt_bf16_kernel<<<dim3((unsigned)((size_t)kV * kD / 1024)), blk256, 0, stream>>>(fc_w, fc_w_bf);

    // ---------------- embedding ----------------
    embed_kernel<<<dim3(kM), blk256, 0, stream>>>(ids, emb, x, x_bf);

    for (int layer = 0; layer < kNL; ++layer) {
        const __bf16* in_w_l  = in_w_bf + (size_t)layer * 2 * kDI * kD;
        const float*  in_b_l  = in_b    + (size_t)layer * 2 * kDI;
        const float*  conv_w_l= conv_w  + (size_t)layer * kDI * kDC;
        const float*  conv_b_l= conv_b  + (size_t)layer * kDI;
        const __bf16* xp_w_l  = xp_w_bf + (size_t)layer * (kDR + 2 * kDS) * kDI;
        const __bf16* dt_w_l  = dt_w_bf + (size_t)layer * kDI * kDR;
        const float*  dt_b_l  = dt_b    + (size_t)layer * kDI;
        const float*  A_log_l = A_log   + (size_t)layer * kDI * kDS;
        const float*  Dp_l    = Dp      + (size_t)layer * kDI;
        const __bf16* ou_w_l  = ou_w_bf + (size_t)layer * kD * kDI;
        const float*  out_b_l = out_b   + (size_t)layer * kD;
        const float*  ln_g_l  = ln_g    + (size_t)layer * kD;
        const float*  ln_b_l  = ln_b    + (size_t)layer * kD;

        // in_proj: xz = x @ in_w^T + in_b   (4096 x 4096, K=1024)
        gemm_bf16_wmma<<<dim3(2 * kDI / 64, kM / 128), blk256, 0, stream>>>(
            x_bf, in_w_l, in_b_l, xz, nullptr, 2 * kDI, kD, kD, kD, 2 * kDI, 0);

        // causal depthwise conv + SiLU -> xi (+bf16)
        conv_silu_kernel<<<gEl, blk256, 0, stream>>>(xz, conv_w_l, conv_b_l, xi, xi_bf);

        // xproj: dbl = xi @ xproj_w^T   (4096 x 80, K=2048)
        gemm_bf16_wmma<<<dim3(2, kM / 128), blk256, 0, stream>>>(
            xi_bf, xp_w_l, nullptr, dbl, dbl_bf, kDR + 2 * kDS, kDI,
            kDI, kDI, kDR + 2 * kDS, 0);

        // dt_proj + softplus: delta = softplus(dt @ dt_w^T + dt_b)  (K=64)
        gemm_bf16_wmma<<<dim3(kDI / 64, kM / 128), blk256, 0, stream>>>(
            dbl_bf, dt_w_l, dt_b_l, delta, nullptr, kDI, kDR,
            kDR + 2 * kDS, kDR, kDI, 1);

        // selective scan -> ysb
        scan_kernel<<<dim3(kB * kDI / 32), blk256, 0, stream>>>(
            delta, xi, dbl, A_log_l, ysb);

        // gating: ysb = (ysb + xi*Dp) * silu(z)  (+bf16)
        gate_kernel<<<gEl, blk256, 0, stream>>>(ysb, ys_bf, xi, xz, Dp_l);

        // out_proj: outb = ysb @ out_w^T + out_b  (4096 x 1024, K=2048)
        gemm_bf16_wmma<<<dim3(kD / 64, kM / 128), blk256, 0, stream>>>(
            ys_bf, ou_w_l, out_b_l, outb, nullptr, kD, kDI, kDI, kDI, kD, 0);

        // residual + LayerNorm: x = LN(outb + x)  (f32 + bf16 for next GEMM)
        ln_kernel<<<dim3(kM), blk256, 0, stream>>>(outb, x, ln_g_l, ln_b_l, x, x_bf);
    }

    // final LayerNorm -> bf16 only (fc consumes it)
    ln_kernel<<<dim3(kM), blk256, 0, stream>>>(x, nullptr, fn_g, fn_b, nullptr, xn_bf);

    // vocab head: logits = xn @ fc_w^T + fc_b  (4096 x 32000, K=1024)
    gemm_bf16_wmma<<<dim3(kV / 64, kM / 128), blk256, 0, stream>>>(
        xn_bf, fc_w_bf, fc_b, logits, nullptr, kV, kD, kD, kD, kV, 0);
}
